// TransformerDecoder_62457414418616
// MI455X (gfx1250) — matrix operand order repaired and checked by reading, not compiled, not run on hardware
//
#include <hip/hip_runtime.h>
#include <cstddef>

// ---------------- constants for this problem instance ----------------
#define Bc     2
#define NQc    2048
#define NVc    2048
#define DIMc   256
#define HEADSc 8
#define HDc    32
#define HIDc   1024
#define Kc     8
#define Mrows  (Bc * NQc)         // 4096 token rows
#define SCALEc 0.17677669529663687f  // 32^-0.5

typedef __attribute__((ext_vector_type(16))) __bf16 bf16x16;
typedef __attribute__((ext_vector_type(8)))  float  f32x8;

#define DEV static __device__ __forceinline__

// round-to-nearest-even f32 -> bf16 bits
DEV unsigned short f2b(float f) {
  unsigned u = __float_as_uint(f);
  u += 0x7fffu + ((u >> 16) & 1u);
  return (unsigned short)(u >> 16);
}

DEV bf16x16 mk16(uint4 a, uint4 b) {
  union { unsigned u[8]; bf16x16 v; } t;
  t.u[0]=a.x; t.u[1]=a.y; t.u[2]=a.z; t.u[3]=a.w;
  t.u[4]=b.x; t.u[5]=b.y; t.u[6]=b.z; t.u[7]=b.w;
  return t.v;
}

// A fragment from per-lane base pointer + compile-time-friendly k offset.
// ISA layout: lanes 0-15 -> M=lane, K = {0..7, 16..23}; lanes 16-31 -> K = {8..15, 24..31}
DEV bf16x16 lda_k(const unsigned short* p, int k) {
  return mk16(*(const uint4*)(p + k), *(const uint4*)(p + k + 16));
}
// B fragment (from B^T stored N x K): lanes 0-15 K = {0..7, 16..23}; lanes 16-31 K = {8..15, 24..31}
DEV bf16x16 ldb_k(const unsigned short* p, int k) {
  return mk16(*(const uint4*)(p + k), *(const uint4*)(p + k + 16));
}

// generic-stride loaders (used by attention where strides are runtime)
DEV bf16x16 load_a_g(const unsigned short* A, int lda, int m0, int k0, int lane) {
  int m  = m0 + (lane & 15);
  int kb = k0 + ((lane >> 4) << 3);
  const unsigned short* r = A + (size_t)m * lda + kb;
  return mk16(*(const uint4*)r, *(const uint4*)(r + 16));
}
DEV bf16x16 load_b_g(const unsigned short* BT, int ldbt, int n0, int k0, int lane) {
  int n  = n0 + (lane & 15);
  int kb = k0 + ((lane >> 4) << 3);
  const unsigned short* r = BT + (size_t)n * ldbt + kb;
  return mk16(*(const uint4*)r, *(const uint4*)(r + 16));
}

DEV f32x8 wmma_bf(bf16x16 a, bf16x16 b, f32x8 c) {
  return __builtin_amdgcn_wmma_f32_16x16x32_bf16(false, a, false, b, (short)0, c, false, false);
}

// C layout: lanes 0-15 -> row = mt+i, col = nt+lane; lanes 16-31 -> row = mt+8+i
DEV void store_c(float* C, int ldc, int mt, int nt, f32x8 c,
                 const float* bias, const float* res, int lane) {
  int col   = nt + (lane & 15);
  int rbase = mt + ((lane >> 4) << 3);
  float bv = bias ? bias[col] : 0.0f;
#pragma unroll
  for (int i = 0; i < 8; ++i) {
    size_t off = (size_t)(rbase + i) * ldc + col;
    float v = c[i] + bv;
    if (res) v += res[off];
    C[off] = v;
  }
}

// ---------------- GEMM: C[M,N] = A[M,K](bf16) * BT[N,K](bf16)^T + bias + res ----
// block = 256 threads (8 waves, 4x2), block tile 128x128, wave tile 32x64.
// KD is a template constant (256 or 1024): all loads are base + constant offset,
// so the loop body has no address arithmetic; last k-step is peeled; next-step
// fragments load into fresh registers (no WMMA operand WAR hazard NOPs).
template <int KD>
__global__ __launch_bounds__(256) void gemm_bf16_nt(
    float* __restrict__ C, const unsigned short* __restrict__ A,
    const unsigned short* __restrict__ BT,
    const float* __restrict__ bias, const float* __restrict__ res, int N) {
  int lane = threadIdx.x & 31, w = threadIdx.x >> 5;
  int m0 = blockIdx.y * 128 + (w & 3) * 32;
  int n0 = blockIdx.x * 128 + (w >> 2) * 64;
  const unsigned short* pa0 = A  + (size_t)(m0 + (lane & 15)) * KD + ((lane >> 4) << 3);
  const unsigned short* pa1 = pa0 + (size_t)16 * KD;
  const unsigned short* pb0 = BT + (size_t)(n0 + (lane & 15)) * KD + ((lane >> 4) << 3);
  const unsigned short* pb1 = pb0 + (size_t)16 * KD;
  const unsigned short* pb2 = pb0 + (size_t)32 * KD;
  const unsigned short* pb3 = pb0 + (size_t)48 * KD;

  f32x8 acc[2][4];
#pragma unroll
  for (int i = 0; i < 2; ++i)
#pragma unroll
    for (int j = 0; j < 4; ++j) acc[i][j] = f32x8{0,0,0,0,0,0,0,0};

  bf16x16 a0 = lda_k(pa0, 0), a1 = lda_k(pa1, 0);
  bf16x16 b0 = ldb_k(pb0, 0), b1 = ldb_k(pb1, 0);
  bf16x16 b2 = ldb_k(pb2, 0), b3 = ldb_k(pb3, 0);

#pragma unroll 2
  for (int k = 32; k < KD; k += 32) {
    bf16x16 na0 = lda_k(pa0, k), na1 = lda_k(pa1, k);
    bf16x16 nb0 = ldb_k(pb0, k), nb1 = ldb_k(pb1, k);
    bf16x16 nb2 = ldb_k(pb2, k), nb3 = ldb_k(pb3, k);
    __builtin_prefetch(pa0 + k + 32, 0, 3);   // global_prefetch_b8, OOB dropped
    __builtin_prefetch(pb0 + k + 32, 0, 3);
    acc[0][0] = wmma_bf(a0, b0, acc[0][0]);
    acc[1][0] = wmma_bf(a1, b0, acc[1][0]);
    acc[0][1] = wmma_bf(a0, b1, acc[0][1]);
    acc[1][1] = wmma_bf(a1, b1, acc[1][1]);
    acc[0][2] = wmma_bf(a0, b2, acc[0][2]);
    acc[1][2] = wmma_bf(a1, b2, acc[1][2]);
    acc[0][3] = wmma_bf(a0, b3, acc[0][3]);
    acc[1][3] = wmma_bf(a1, b3, acc[1][3]);
    a0 = na0; a1 = na1; b0 = nb0; b1 = nb1; b2 = nb2; b3 = nb3;
  }
  // peeled final k-step
  acc[0][0] = wmma_bf(a0, b0, acc[0][0]);
  acc[1][0] = wmma_bf(a1, b0, acc[1][0]);
  acc[0][1] = wmma_bf(a0, b1, acc[0][1]);
  acc[1][1] = wmma_bf(a1, b1, acc[1][1]);
  acc[0][2] = wmma_bf(a0, b2, acc[0][2]);
  acc[1][2] = wmma_bf(a1, b2, acc[1][2]);
  acc[0][3] = wmma_bf(a0, b3, acc[0][3]);
  acc[1][3] = wmma_bf(a1, b3, acc[1][3]);

#pragma unroll
  for (int i = 0; i < 2; ++i)
#pragma unroll
    for (int j = 0; j < 4; ++j)
      store_c(C, N, m0 + 16 * i, n0 + 16 * j, acc[i][j], bias, res, lane);
}

// ---------------- Flash attention: per wave 16 queries, hd=32 -------------------
// Q: (tok, qstride) bf16 row-major, head offset h*32
// Km: (tok, kstride) bf16 row-major, offset koff + h*32 (rows = keys => B^T ready)
// VT: head-major transposed V: [((b*8+h)*32 + d) * NKt + key]
// O: f32 (tok, 256), head offset h*32
__global__ __launch_bounds__(128) void attn_fwd(
    float* __restrict__ O, const unsigned short* __restrict__ Q,
    const unsigned short* __restrict__ Km, const unsigned short* __restrict__ VT,
    int NQt, int NKt, int qstride, int kstride, int koff, float scale) {
  __shared__ __align__(16) unsigned short plds[4][16 * 32];
  int lane = threadIdx.x & 31, w = threadIdx.x >> 5;
  int bh = blockIdx.y, b = bh >> 3, h = bh & 7;
  int q0 = blockIdx.x * 64 + w * 16;
  const unsigned short* qp = Q  + (size_t)b * NQt * qstride + h * 32;
  const unsigned short* kp = Km + (size_t)b * NKt * kstride + koff + h * 32;
  const unsigned short* vp = VT + (size_t)bh * 32 * NKt;
  float* op = O + (size_t)b * NQt * 256 + h * 32;

  bf16x16 aq = load_a_g(qp, qstride, q0, 0, lane);
  f32x8 o0 = {0,0,0,0,0,0,0,0}, o1 = {0,0,0,0,0,0,0,0};
  float mrow[8], lrow[8];
#pragma unroll
  for (int i = 0; i < 8; ++i) { mrow[i] = -1e30f; lrow[i] = 0.0f; }
  int hi8 = (lane >> 4) << 3;
  unsigned short* pl = plds[w];

  for (int kt = 0; kt < NKt; kt += 32) {
    bf16x16 kf0 = load_b_g(kp, kstride, kt,      0, lane);
    bf16x16 kf1 = load_b_g(kp, kstride, kt + 16, 0, lane);
    f32x8 z = {0,0,0,0,0,0,0,0};
    f32x8 s0 = wmma_bf(aq, kf0, z);
    f32x8 s1 = wmma_bf(aq, kf1, z);
#pragma unroll
    for (int i = 0; i < 8; ++i) {
      float x0 = s0[i] * scale, x1 = s1[i] * scale;
      float mx = fmaxf(x0, x1);
      mx = fmaxf(mx, __shfl_xor(mx, 1));
      mx = fmaxf(mx, __shfl_xor(mx, 2));
      mx = fmaxf(mx, __shfl_xor(mx, 4));
      mx = fmaxf(mx, __shfl_xor(mx, 8));   // row max over 16-lane half group
      float nm = fmaxf(mrow[i], mx);
      float al = __expf(mrow[i] - nm);
      float p0 = __expf(x0 - nm), p1 = __expf(x1 - nm);
      mrow[i] = nm;
      float rs = p0 + p1;
      rs += __shfl_xor(rs, 1); rs += __shfl_xor(rs, 2);
      rs += __shfl_xor(rs, 4); rs += __shfl_xor(rs, 8);
      lrow[i] = lrow[i] * al + rs;
      o0[i] *= al;  o1[i] *= al;
      int r = i + hi8;                       // C-layout row
      pl[r * 32 + (lane & 15)]      = f2b(p0);
      pl[r * 32 + 16 + (lane & 15)] = f2b(p1);
    }
    __syncthreads();
    {
      int m = lane & 15;                     // A-layout re-read of P from LDS
      const unsigned short* rp = pl + m * 32 + hi8;
      bf16x16 pa = mk16(*(const uint4*)rp, *(const uint4*)(rp + 16));
      bf16x16 vf0 = load_b_g(vp, NKt, 0,  kt, lane);
      bf16x16 vf1 = load_b_g(vp, NKt, 16, kt, lane);
      o0 = wmma_bf(pa, vf0, o0);
      o1 = wmma_bf(pa, vf1, o1);
    }
    __syncthreads();
  }
#pragma unroll
  for (int i = 0; i < 8; ++i) {
    int r = q0 + i + hi8;
    float inv = 1.0f / lrow[i];
    op[(size_t)r * 256 + (lane & 15)]      = o0[i] * inv;
    op[(size_t)r * 256 + 16 + (lane & 15)] = o1[i] * inv;
  }
}

// ---------------- LayerNorm over rows of 256, emit bf16 -------------------------
__global__ __launch_bounds__(256) void ln_rows(
    const float* __restrict__ X, const float* __restrict__ G,
    const float* __restrict__ Bt, unsigned short* __restrict__ Y) {
  int row = blockIdx.x, tid = threadIdx.x;
  float x = X[(size_t)row * 256 + tid];
  float s = x, s2 = x * x;
  for (int off = 16; off; off >>= 1) { s += __shfl_xor(s, off); s2 += __shfl_xor(s2, off); }
  __shared__ float sh[16];
  int wv = tid >> 5, lane = tid & 31;
  if (lane == 0) { sh[wv] = s; sh[wv + 8] = s2; }
  __syncthreads();
  float ts = 0, ts2 = 0;
#pragma unroll
  for (int i = 0; i < 8; ++i) { ts += sh[i]; ts2 += sh[i + 8]; }
  float mean = ts * (1.0f / 256.0f);
  float var  = ts2 * (1.0f / 256.0f) - mean * mean;
  float y = (x - mean) * rsqrtf(var + 1e-5f) * G[tid] + Bt[tid];
  Y[(size_t)row * 256 + tid] = f2b(y);
}

// ---------------- small helper kernels ------------------------------------------
__global__ void wt_convert(const float* __restrict__ W, unsigned short* __restrict__ T,
                           int Kd, int N) {            // W: KxN -> T: NxK (bf16)
  int idx = blockIdx.x * 256 + threadIdx.x;
  int k = idx / N, n = idx % N;
  T[(size_t)n * Kd + k] = f2b(W[idx]);
}

// W: (2K x N). T1 = top^T; T2 = (diffmode ? bot-top : bot)^T
__global__ void wt_split2(const float* __restrict__ W, unsigned short* __restrict__ T1,
                          unsigned short* __restrict__ T2, int Kd, int N, int diffmode) {
  int idx = blockIdx.x * 256 + threadIdx.x;
  int k = idx / N, n = idx % N;
  float top = W[(size_t)k * N + n];
  float bot = W[(size_t)(k + Kd) * N + n];
  T1[(size_t)n * Kd + k] = f2b(top);
  T2[(size_t)n * Kd + k] = f2b(diffmode ? (bot - top) : bot);
}

__global__ void conv_bf16(const float* __restrict__ X, unsigned short* __restrict__ Y, int n) {
  int i = blockIdx.x * 256 + threadIdx.x;
  if (i < n) Y[i] = f2b(X[i]);
}

__global__ void gelu_bf16(const float* __restrict__ X, unsigned short* __restrict__ Y, int n) {
  int i = blockIdx.x * 256 + threadIdx.x;
  if (i < n) {
    float x = X[i];
    Y[i] = f2b(0.5f * x * (1.0f + erff(x * 0.7071067811865475f)));
  }
}

__global__ void add_f32(float* __restrict__ O, const float* __restrict__ A,
                        const float* __restrict__ Bp, int n) {
  int i = blockIdx.x * 256 + threadIdx.x;
  if (i < n) O[i] = A[i] + Bp[i];
}

// V transpose-convert: src (B*Ntok x sstride f32) -> dst[((b*8+h)*32+d)*Ntok + t]
__global__ void vt_convert(const float* __restrict__ S, unsigned short* __restrict__ D,
                           int Ntok, int sstride, int soff) {
  int idx = blockIdx.x * 256 + threadIdx.x;
  int t = idx % Ntok;
  int r = idx / Ntok;
  int d = r & 31, h = (r >> 5) & 7, b = r >> 8;
  D[idx] = f2b(S[(size_t)(b * Ntok + t) * sstride + soff + h * 32 + d]);
}

// knn: Y[row,c] = max_k lrelu(G[row,c] + H[b*Nv + idx[row,k], c]), bf16 out
__global__ __launch_bounds__(256) void knn_reduce(
    const float* __restrict__ Hm, const float* __restrict__ G,
    const int* __restrict__ I, unsigned short* __restrict__ Y, int Nv) {
  int row = blockIdx.x, tid = threadIdx.x;
  __shared__ int sidx[Kc];
  if (tid < Kc) sidx[tid] = I[(size_t)row * Kc + tid];
  __syncthreads();
  int b = row / NQc;
  float g = G[(size_t)row * 256 + tid];
  float m = -1e30f;
#pragma unroll
  for (int k = 0; k < Kc; ++k) {
    float v = g + Hm[(size_t)(b * Nv + sidx[k]) * 256 + tid];
    v = v > 0.0f ? v : 0.2f * v;
    m = fmaxf(m, v);
  }
  Y[(size_t)row * 256 + tid] = f2b(m);
}

// =====================================================================================
extern "C" void kernel_launch(void* const* d_in, const int* in_sizes, int n_in,
                              void* d_out, int out_size, void* d_ws, size_t ws_size,
                              hipStream_t stream) {
  (void)in_sizes; (void)n_in; (void)out_size; (void)ws_size;
  const float* q_in   = (const float*)d_in[0];
  const float* v_in   = (const float*)d_in[1];
  const int*   selfI  = (const int*)d_in[4];
  const int*   crossI = (const int*)d_in[5];
  const float *n1g = (const float*)d_in[6],  *n1b = (const float*)d_in[7];
  const float *nqg = (const float*)d_in[8],  *nqb = (const float*)d_in[9];
  const float *nvg = (const float*)d_in[10], *nvb = (const float*)d_in[11];
  const float *n2g = (const float*)d_in[12], *n2b = (const float*)d_in[13];
  const float* sa_qkv_w  = (const float*)d_in[14];
  const float* sa_proj_w = (const float*)d_in[15];
  const float* sa_proj_b = (const float*)d_in[16];
  const float* ca_q_w    = (const float*)d_in[17];
  const float* ca_kv_w   = (const float*)d_in[18];
  const float* ca_proj_w = (const float*)d_in[19];
  const float* ca_proj_b = (const float*)d_in[20];
  const float* fc1_w = (const float*)d_in[21];
  const float* fc1_b = (const float*)d_in[22];
  const float* fc2_w = (const float*)d_in[23];
  const float* fc2_b = (const float*)d_in[24];
  const float* knn_w = (const float*)d_in[25];
  const float* knn_b = (const float*)d_in[26];
  const float* merge_w = (const float*)d_in[27];
  const float* merge_b = (const float*)d_in[28];
  const float* knnx_w = (const float*)d_in[29];
  const float* knnx_b = (const float*)d_in[30];
  const float* mergex_w = (const float*)d_in[31];
  const float* mergex_b = (const float*)d_in[32];
  float* out = (float*)d_out;

  // ---- workspace carve-out (256B aligned) ----
  char* wp = (char*)d_ws;
  auto alloc = [&](size_t bytes) -> void* {
    void* r = (void*)wp; wp += (bytes + 255) & ~(size_t)255; return r;
  };
  typedef unsigned short u16;
  const size_t T = Mrows;                 // 4096 token rows
  // bf16 weights (transposed N x K)
  u16* w_qkv   = (u16*)alloc(768 * 256 * 2);
  u16* w_sapr  = (u16*)alloc(256 * 256 * 2);
  u16* w_caq   = (u16*)alloc(256 * 256 * 2);
  u16* w_cakv  = (u16*)alloc(512 * 256 * 2);
  u16* w_capr  = (u16*)alloc(256 * 256 * 2);
  u16* w_fc1   = (u16*)alloc(1024 * 256 * 2);
  u16* w_fc2   = (u16*)alloc(256 * 1024 * 2);
  u16* w_ktop  = (u16*)alloc(256 * 256 * 2);
  u16* w_kdif  = (u16*)alloc(256 * 256 * 2);
  u16* w_xtop  = (u16*)alloc(256 * 256 * 2);
  u16* w_xdif  = (u16*)alloc(256 * 256 * 2);
  u16* w_mtop  = (u16*)alloc(256 * 256 * 2);
  u16* w_mbot  = (u16*)alloc(256 * 256 * 2);
  u16* w_mxtop = (u16*)alloc(256 * 256 * 2);
  u16* w_mxbot = (u16*)alloc(256 * 256 * 2);
  // bf16 activations
  u16* nq_b   = (u16*)alloc(T * 256 * 2);
  u16* qkv_b  = (u16*)alloc(T * 768 * 2);
  u16* vt_b   = (u16*)alloc((size_t)Bc * 8 * 32 * NQc * 2);
  u16* o_b    = (u16*)alloc(T * 256 * 2);
  u16* pr_b   = (u16*)alloc(T * 256 * 2);   // proj output (sa_out / ca_out) bf16
  u16* kn_b   = (u16*)alloc(T * 256 * 2);   // knn max output bf16
  u16* nq2_b  = (u16*)alloc(T * 256 * 2);
  u16* nv_b   = (u16*)alloc(T * 256 * 2);
  u16* qc_b   = (u16*)alloc(T * 256 * 2);
  u16* kv_b   = (u16*)alloc(T * 512 * 2);
  u16* vtx_b  = (u16*)alloc((size_t)Bc * 8 * 32 * NVc * 2);
  u16* h2_b   = (u16*)alloc(T * 256 * 2);
  u16* hid_b  = (u16*)alloc(T * 1024 * 2);
  // f32 scratch
  float* fbig  = (float*)alloc(T * 1024 * 4); // qkv_f then hid_f
  float* ao_f  = (float*)alloc(T * 256 * 4);  // attention out f32
  float* pr_f  = (float*)alloc(T * 256 * 4);  // proj out f32
  float* h_f   = (float*)alloc(T * 256 * 4);
  float* g_f   = (float*)alloc(T * 256 * 4);
  float* mrg_f = (float*)alloc(T * 256 * 4);
  float* q2_f  = (float*)alloc(T * 256 * 4);
  float* kv_f  = (float*)alloc(T * 512 * 4);

  dim3 blk256(256), blk128(128);
  auto gGemm = [](int N) { return dim3(N / 128, Mrows / 128); };
  dim3 gAttn(NQc / 64, Bc * HEADSc);
  int E256 = T * 256, E768 = T * 768, E512 = T * 512, E1024 = T * 1024;

  // ---- weight prep ----
  wt_convert<<<768 * 256 / 256, blk256, 0, stream>>>(sa_qkv_w, w_qkv, 256, 768);
  wt_convert<<<256 * 256 / 256, blk256, 0, stream>>>(sa_proj_w, w_sapr, 256, 256);
  wt_convert<<<256 * 256 / 256, blk256, 0, stream>>>(ca_q_w, w_caq, 256, 256);
  wt_convert<<<512 * 256 / 256, blk256, 0, stream>>>(ca_kv_w, w_cakv, 256, 512);
  wt_convert<<<256 * 256 / 256, blk256, 0, stream>>>(ca_proj_w, w_capr, 256, 256);
  wt_convert<<<1024 * 256 / 256, blk256, 0, stream>>>(fc1_w, w_fc1, 256, 1024);
  wt_convert<<<1024 * 256 / 256, blk256, 0, stream>>>(fc2_w, w_fc2, 1024, 256);
  wt_split2<<<256 * 256 / 256, blk256, 0, stream>>>(knn_w,   w_ktop,  w_kdif,  256, 256, 1);
  wt_split2<<<256 * 256 / 256, blk256, 0, stream>>>(knnx_w,  w_xtop,  w_xdif,  256, 256, 1);
  wt_split2<<<256 * 256 / 256, blk256, 0, stream>>>(merge_w, w_mtop,  w_mbot,  256, 256, 0);
  wt_split2<<<256 * 256 / 256, blk256, 0, stream>>>(mergex_w,w_mxtop, w_mxbot, 256, 256, 0);

  // ================= Block 1: self-attn + knn =================
  ln_rows<<<Mrows, blk256, 0, stream>>>(q_in, n1g, n1b, nq_b);
  gemm_bf16_nt<256><<<gGemm(768), blk256, 0, stream>>>(fbig, nq_b, w_qkv, nullptr, nullptr, 768);
  conv_bf16<<<E768 / 256, blk256, 0, stream>>>(fbig, qkv_b, E768);
  vt_convert<<<(Bc * 8 * 32 * NQc) / 256, blk256, 0, stream>>>(fbig, vt_b, NQc, 768, 512);
  attn_fwd<<<gAttn, blk128, 0, stream>>>(ao_f, qkv_b, qkv_b, vt_b,
                                         NQc, NQc, 768, 768, 256, SCALEc);
  conv_bf16<<<E256 / 256, blk256, 0, stream>>>(ao_f, o_b, E256);
  gemm_bf16_nt<256><<<gGemm(256), blk256, 0, stream>>>(pr_f, o_b, w_sapr, sa_proj_b, nullptr, 256);
  conv_bf16<<<E256 / 256, blk256, 0, stream>>>(pr_f, pr_b, E256);
  // knn: h = nq @ W_top ; g = nq @ (W_bot - W_top) + b
  gemm_bf16_nt<256><<<gGemm(256), blk256, 0, stream>>>(h_f, nq_b, w_ktop, nullptr, nullptr, 256);
  gemm_bf16_nt<256><<<gGemm(256), blk256, 0, stream>>>(g_f, nq_b, w_kdif, knn_b, nullptr, 256);
  knn_reduce<<<Mrows, blk256, 0, stream>>>(h_f, g_f, selfI, kn_b, NQc);
  // merge: mrg = sa_out @ mtop + merge_b ; mrg += knn_out @ mbot
  gemm_bf16_nt<256><<<gGemm(256), blk256, 0, stream>>>(mrg_f, pr_b, w_mtop, merge_b, nullptr, 256);
  gemm_bf16_nt<256><<<gGemm(256), blk256, 0, stream>>>(mrg_f, kn_b, w_mbot, nullptr, mrg_f, 256);
  add_f32<<<E256 / 256, blk256, 0, stream>>>(q2_f, q_in, mrg_f, E256);

  // ================= Block 2: cross-attn + knnx =================
  ln_rows<<<Mrows, blk256, 0, stream>>>(q2_f, nqg, nqb, nq2_b);
  ln_rows<<<Mrows, blk256, 0, stream>>>(v_in, nvg, nvb, nv_b);
  gemm_bf16_nt<256><<<gGemm(256), blk256, 0, stream>>>(pr_f, nq2_b, w_caq, nullptr, nullptr, 256);
  conv_bf16<<<E256 / 256, blk256, 0, stream>>>(pr_f, qc_b, E256);
  gemm_bf16_nt<256><<<gGemm(512), blk256, 0, stream>>>(kv_f, nv_b, w_cakv, nullptr, nullptr, 512);
  conv_bf16<<<E512 / 256, blk256, 0, stream>>>(kv_f, kv_b, E512);
  vt_convert<<<(Bc * 8 * 32 * NVc) / 256, blk256, 0, stream>>>(kv_f, vtx_b, NVc, 512, 256);
  attn_fwd<<<gAttn, blk128, 0, stream>>>(ao_f, qc_b, kv_b, vtx_b,
                                         NQc, NVc, 256, 512, 0, SCALEc);
  conv_bf16<<<E256 / 256, blk256, 0, stream>>>(ao_f, o_b, E256);
  gemm_bf16_nt<256><<<gGemm(256), blk256, 0, stream>>>(pr_f, o_b, w_capr, ca_proj_b, nullptr, 256);
  conv_bf16<<<E256 / 256, blk256, 0, stream>>>(pr_f, pr_b, E256);
  gemm_bf16_nt<256><<<gGemm(256), blk256, 0, stream>>>(h_f, nv_b,  w_xtop, nullptr, nullptr, 256);
  gemm_bf16_nt<256><<<gGemm(256), blk256, 0, stream>>>(g_f, nq2_b, w_xdif, knnx_b, nullptr, 256);
  knn_reduce<<<Mrows, blk256, 0, stream>>>(h_f, g_f, crossI, kn_b, NVc);
  gemm_bf16_nt<256><<<gGemm(256), blk256, 0, stream>>>(mrg_f, pr_b, w_mxtop, mergex_b, nullptr, 256);
  gemm_bf16_nt<256><<<gGemm(256), blk256, 0, stream>>>(mrg_f, kn_b, w_mxbot, nullptr, mrg_f, 256);
  add_f32<<<E256 / 256, blk256, 0, stream>>>(out, q2_f, mrg_f, E256);   // q3 lives in d_out

  // ================= Block 3: MLP =================
  ln_rows<<<Mrows, blk256, 0, stream>>>(out, n2g, n2b, h2_b);
  gemm_bf16_nt<256><<<gGemm(1024), blk256, 0, stream>>>(fbig, h2_b, w_fc1, fc1_b, nullptr, 1024);
  gelu_bf16<<<E1024 / 256, blk256, 0, stream>>>(fbig, hid_b, E1024);
  gemm_bf16_nt<1024><<<gGemm(256), blk256, 0, stream>>>(out, hid_b, w_fc2, fc2_b, out, 256);
}
